// BidirectionalMambaBlock_12369505812975
// MI455X (gfx1250) — matrix operand order repaired and checked by reading, not compiled, not run on hardware
//
#include <hip/hip_runtime.h>
#include <math.h>

typedef _Float16 f16;
typedef __attribute__((ext_vector_type(16))) _Float16 v16h;
typedef __attribute__((ext_vector_type(8)))  float    v8f;

#define BB   2
#define LL   2048
#define DM   768
#define DI   1536
#define NS   16
#define NPROJ 33          // 1 + 2*NS
#define MTOT (BB*LL)      // 4096
#define NIN  (2*DI)       // 3072

#if defined(__has_builtin)
# if __has_builtin(__builtin_amdgcn_global_load_async_to_lds_b32) && \
     __has_builtin(__builtin_amdgcn_s_wait_asynccnt)
#  define USE_ASYNC_LDS 1
# endif
#endif

typedef __attribute__((address_space(1))) int gint;
typedef __attribute__((address_space(3))) int lint;

// ---------------------------------------------------------------------------
// LayerNorm: one block per row (768 cols). Writes fp32 and f16 copies.
// ---------------------------------------------------------------------------
__global__ void __launch_bounds__(256)
ln_kernel(const float* __restrict__ x, const float* __restrict__ g,
          const float* __restrict__ bt, float* __restrict__ xn,
          f16* __restrict__ xn16)
{
    int m = blockIdx.x;
    const float* row = x + (size_t)m * DM;
    __shared__ float ssum[256], ssq[256];
    float s = 0.f, s2 = 0.f;
    for (int i = threadIdx.x; i < DM; i += 256) {
        float v = row[i]; s += v; s2 += v * v;
    }
    ssum[threadIdx.x] = s; ssq[threadIdx.x] = s2;
    __syncthreads();
    for (int st = 128; st > 0; st >>= 1) {
        if (threadIdx.x < st) {
            ssum[threadIdx.x] += ssum[threadIdx.x + st];
            ssq[threadIdx.x]  += ssq[threadIdx.x + st];
        }
        __syncthreads();
    }
    float mu  = ssum[0] * (1.f / DM);
    float var = ssq[0] * (1.f / DM) - mu * mu;
    float rs  = rsqrtf(var + 1e-5f);
    for (int i = threadIdx.x; i < DM; i += 256) {
        float v = (row[i] - mu) * rs * g[i] + bt[i];
        xn[(size_t)m * DM + i]   = v;
        xn16[(size_t)m * DM + i] = (f16)v;
    }
}

// ---------------------------------------------------------------------------
// Pack fp32 weight [K x N] into per-lane WMMA B fragments (f16):
// packed[((kt*(N/16)+nt)*32 + lane)*16 + i]
//   lane<16 : col = nt*16+lane,    K = kt*32 + (i<8 ? i : 16 + i-8)
//   lane>=16: col = nt*16+lane-16, K = kt*32 + (i<8 ? 8 + i : 24 + i-8)
// ---------------------------------------------------------------------------
__global__ void pack_w_kernel(const float* __restrict__ w, f16* __restrict__ p,
                              int K, int N)
{
    size_t total = (size_t)K * N;
    for (size_t idx = (size_t)blockIdx.x * blockDim.x + threadIdx.x;
         idx < total; idx += (size_t)gridDim.x * blockDim.x) {
        size_t t = idx;
        int i    = t & 15;  t >>= 4;
        int lane = t & 31;  t >>= 5;
        int nt   = (int)(t % (N / 16));
        int kt   = (int)(t / (N / 16));
        int n = nt * 16 + (lane & 15);
        int k = kt * 32 + ((lane < 16) ? 0 : 8) + ((i < 8) ? i : i + 8);
        p[idx] = (f16)w[(size_t)k * N + n];
    }
}

// ---------------------------------------------------------------------------
// WMMA in-projection: xz[m, 0..3071] = xn_row(dir,m) @ inw  (f16 in, f32 out)
// Each wave: 16x64 strip. Double-buffered K loop (64 per outer iteration)
// so fragment loads for step k+1 overlap the WMMAs of step k.
// ---------------------------------------------------------------------------
__global__ void __launch_bounds__(256)
gemm_in_kernel(const f16* __restrict__ xn16, const f16* __restrict__ wpack,
               float* __restrict__ xz, int dir)
{
    const int K = DM, N = NIN;
    int lane = threadIdx.x & 31;
    int wave = threadIdx.x >> 5;
    int m0 = blockIdx.y * 16;
    int n0 = (blockIdx.x * 8 + wave) * 64;
    if (n0 >= N) return;

    int mrow = m0 + (lane & 15);
    int b = mrow / LL, l = mrow % LL;
    int arow = dir ? (b * LL + (LL - 1 - l)) : mrow;
    const f16* aptr = xn16 + (size_t)arow * K;
    int koff = (lane < 16) ? 0 : 8;
    const int ntiles = N / 16;
    const v16h* bcol = (const v16h*)wpack + (size_t)(n0 >> 4) * 32 + lane;

    union AF { v16h v; unsigned u[8]; };
    AF A0, A1;
    v16h B0[4], B1[4];

    auto load_frags = [&](int k0, AF& A, v16h* Bv) {
#pragma unroll
        for (int p = 0; p < 8; ++p) {
            int kk = k0 + koff + ((p < 4) ? 2 * p : 2 * p + 8);
            A.u[p] = *(const unsigned*)(aptr + kk);
        }
        const v16h* bb = bcol + ((size_t)(k0 >> 5) * ntiles) * 32;
#pragma unroll
        for (int t = 0; t < 4; ++t) Bv[t] = bb[(size_t)t * 32];
    };

    v8f c[4] = {};
    load_frags(0, A0, B0);
    for (int k0 = 0; k0 < K; k0 += 64) {
        load_frags(k0 + 32, A1, B1);
        __builtin_prefetch(aptr + k0 + 128, 0, 3);   // global_prefetch_b8 (WGP)
#pragma unroll
        for (int t = 0; t < 4; ++t)
            c[t] = __builtin_amdgcn_wmma_f32_16x16x32_f16(
                       false, A0.v, false, B0[t], (short)0, c[t], false, false);
        if (k0 + 64 < K) load_frags(k0 + 64, A0, B0);
#pragma unroll
        for (int t = 0; t < 4; ++t)
            c[t] = __builtin_amdgcn_wmma_f32_16x16x32_f16(
                       false, A1.v, false, B1[t], (short)0, c[t], false, false);
    }

    int rbase = m0 + ((lane >> 4) << 3);
    int col   = lane & 15;
#pragma unroll
    for (int j = 0; j < 8; ++j) {
        float* orow = xz + (size_t)(rbase + j) * N;
#pragma unroll
        for (int t = 0; t < 4; ++t)
            orow[n0 + t * 16 + col] = c[t][j];
    }
}

// ---------------------------------------------------------------------------
// Depthwise causal conv (k=4) + bias + SiLU on xi (= xz cols [0,1536))
// ---------------------------------------------------------------------------
__global__ void __launch_bounds__(256)
conv_silu_kernel(const float* __restrict__ xz, const float* __restrict__ cw,
                 const float* __restrict__ cb, float* __restrict__ xc)
{
    size_t idx = (size_t)blockIdx.x * blockDim.x + threadIdx.x;
    if (idx >= (size_t)MTOT * DI) return;
    int d = (int)(idx % DI);
    size_t m = idx / DI;
    int b = (int)(m / LL), l = (int)(m % LL);
    float acc = cb[d];
#pragma unroll
    for (int j = 0; j < 4; ++j) {
        int ll = l - 3 + j;
        if (ll >= 0)
            acc += cw[d * 4 + j] * xz[((size_t)(b * LL + ll)) * NIN + d];
    }
    float sg = 1.f / (1.f + __expf(-acc));
    xc[idx] = acc * sg;
}

// ---------------------------------------------------------------------------
// Small projection: bcdt[m, 0..32] = xc_row @ xw.  Row staged into LDS via
// ASYNCcnt-tracked global_load_async_to_lds (when toolchain supports it),
// then a 4-way K-split dot product (33 cols x 4 chunks = 132 active threads).
// ---------------------------------------------------------------------------
__global__ void __launch_bounds__(256)
bcdt_kernel(const float* __restrict__ xc, const float* __restrict__ xw,
            float* __restrict__ bcdt)
{
    __shared__ float row[DI];
    __shared__ float part[4][64];
    size_t m = blockIdx.x;
    const float* src = xc + m * DI;
#ifdef USE_ASYNC_LDS
    for (int i = threadIdx.x; i < DI; i += 256) {
        __builtin_amdgcn_global_load_async_to_lds_b32(
            (gint*)(src + i), (lint*)(&row[i]), 0, 0);
    }
    __builtin_amdgcn_s_wait_asynccnt(0);
#else
    for (int i = threadIdx.x; i < DI; i += 256) row[i] = src[i];
#endif
    __syncthreads();
    int col   = threadIdx.x & 63;
    int chunk = threadIdx.x >> 6;          // 0..3, 384 K each
    if (col < NPROJ) {
        float acc = 0.f;
        int k0 = chunk * (DI / 4);
        for (int k = k0; k < k0 + DI / 4; ++k)
            acc += row[k] * xw[k * NPROJ + col];
        part[chunk][col] = acc;
    }
    __syncthreads();
    if (threadIdx.x < NPROJ) {
        float acc = part[0][threadIdx.x] + part[1][threadIdx.x] +
                    part[2][threadIdx.x] + part[3][threadIdx.x];
        bcdt[m * NPROJ + threadIdx.x] = acc;
    }
}

// ---------------------------------------------------------------------------
// Selective scan: one thread per channel, 16-state h in VGPRs, serial in t.
// Fuses dt=softplus, dA=exp, clip, y = h.C + xc*D, gate by silu(z); f16 out.
// ---------------------------------------------------------------------------
__global__ void __launch_bounds__(256)
scan_kernel(const float* __restrict__ bcdt, const float* __restrict__ xc,
            const float* __restrict__ xz, const float* __restrict__ dtw,
            const float* __restrict__ dtb, const float* __restrict__ Alog,
            const float* __restrict__ Dp, f16* __restrict__ y16)
{
    int b = blockIdx.y;
    int d = blockIdx.x * 256 + threadIdx.x;
    float A[NS], h[NS];
#pragma unroll
    for (int s = 0; s < NS; ++s) {
        float al = fminf(6.f, fmaxf(-6.f, Alog[d * NS + s]));
        A[s] = -__expf(al);
        h[s] = 0.f;
    }
    float wdt = dtw[d], bdt = dtb[d], Dv = Dp[d];
    __shared__ float sB[NS], sC[NS], sdt;
    for (int t = 0; t < LL; ++t) {
        size_t m = (size_t)b * LL + t;
        if (threadIdx.x < NPROJ) {
            float v = bcdt[m * NPROJ + threadIdx.x];
            if (threadIdx.x == 0)       sdt = v;
            else if (threadIdx.x <= NS) sB[threadIdx.x - 1] = v;
            else                        sC[threadIdx.x - 1 - NS] = v;
        }
        __builtin_prefetch(bcdt + (m + 1) * NPROJ, 0, 3);
        __syncthreads();
        float xv  = xc[m * DI + d];
        float dta = sdt * wdt + bdt;
        float dt  = (dta > 20.f) ? dta : log1pf(__expf(dta));
        float y = 0.f;
#pragma unroll
        for (int s = 0; s < NS; ++s) {
            float dA = __expf(dt * A[s]);
            float nh = dA * h[s] + dt * sB[s] * xv;
            nh = fminf(10000.f, fmaxf(-10000.f, nh));
            h[s] = nh;
            y += nh * sC[s];
        }
        y += xv * Dv;
        float z  = xz[m * NIN + DI + d];
        float sz = z / (1.f + __expf(-z));
        y *= sz;
        y16[m * DI + d] = (f16)y;
        __syncthreads();
    }
}

// ---------------------------------------------------------------------------
// WMMA out-projection, fused: out = a*(yf@Wf) + (1-a)*(yb_flip@Wb) + x
// Same double-buffered K loop as gemm_in; two passes (fwd weights, bwd
// weights with time-flipped row indexing folded into the A-row gather).
// ---------------------------------------------------------------------------
__global__ void __launch_bounds__(256)
gemm_out_kernel(const f16* __restrict__ yf, const f16* __restrict__ yb,
                const f16* __restrict__ wf, const f16* __restrict__ wb,
                const float* __restrict__ x, const float* __restrict__ alpha,
                float* __restrict__ out)
{
    const int K = DI, N = DM;
    int lane = threadIdx.x & 31;
    int wave = threadIdx.x >> 5;
    int strip = blockIdx.x * 8 + wave;
    if (strip >= N / 64) return;
    int n0 = strip * 64;
    int m0 = blockIdx.y * 16;

    int mrow = m0 + (lane & 15);
    int b = mrow / LL, l = mrow % LL;
    const f16* arows[2];
    arows[0] = yf + (size_t)mrow * K;
    arows[1] = yb + (size_t)(b * LL + (LL - 1 - l)) * K;
    const f16* wp[2] = { wf, wb };
    int koff = (lane < 16) ? 0 : 8;
    const int ntiles = N / 16;

    union AF { v16h v; unsigned u[8]; };
    v8f acc[2][4];

    for (int dd = 0; dd < 2; ++dd) {
        const f16* aptr = arows[dd];
        const v16h* bcol = (const v16h*)wp[dd] + (size_t)(n0 >> 4) * 32 + lane;
        auto load_frags = [&](int k0, AF& A, v16h* Bv) {
#pragma unroll
            for (int p = 0; p < 8; ++p) {
                int kk = k0 + koff + ((p < 4) ? 2 * p : 2 * p + 8);
                A.u[p] = *(const unsigned*)(aptr + kk);
            }
            const v16h* bb = bcol + ((size_t)(k0 >> 5) * ntiles) * 32;
#pragma unroll
            for (int t = 0; t < 4; ++t) Bv[t] = bb[(size_t)t * 32];
        };

        AF A0, A1;
        v16h B0[4], B1[4];
#pragma unroll
        for (int t = 0; t < 4; ++t) acc[dd][t] = (v8f){};
        load_frags(0, A0, B0);
        for (int k0 = 0; k0 < K; k0 += 64) {
            load_frags(k0 + 32, A1, B1);
            __builtin_prefetch(aptr + k0 + 128, 0, 3);
#pragma unroll
            for (int t = 0; t < 4; ++t)
                acc[dd][t] = __builtin_amdgcn_wmma_f32_16x16x32_f16(
                        false, A0.v, false, B0[t], (short)0, acc[dd][t], false, false);
            if (k0 + 64 < K) load_frags(k0 + 64, A0, B0);
#pragma unroll
            for (int t = 0; t < 4; ++t)
                acc[dd][t] = __builtin_amdgcn_wmma_f32_16x16x32_f16(
                        false, A1.v, false, B1[t], (short)0, acc[dd][t], false, false);
        }
    }

    float a = 1.f / (1.f + __expf(-alpha[0]));
    int rbase = m0 + ((lane >> 4) << 3);
    int col   = lane & 15;
#pragma unroll
    for (int j = 0; j < 8; ++j) {
        const float* xrow = x + (size_t)(rbase + j) * N;
        float* orow = out + (size_t)(rbase + j) * N;
#pragma unroll
        for (int t = 0; t < 4; ++t) {
            int n = n0 + t * 16 + col;
            orow[n] = a * acc[0][t][j] + (1.f - a) * acc[1][t][j] + xrow[n];
        }
    }
}

// ---------------------------------------------------------------------------
extern "C" void kernel_launch(void* const* d_in, const int* in_sizes, int n_in,
                              void* d_out, int out_size, void* d_ws, size_t ws_size,
                              hipStream_t stream)
{
    (void)in_sizes; (void)n_in; (void)out_size; (void)ws_size;
    const float* x     = (const float*)d_in[0];
    const float* ln_g  = (const float*)d_in[1];
    const float* ln_b  = (const float*)d_in[2];
    const float* alpha = (const float*)d_in[3];
    const float* inw[2]  = { (const float*)d_in[4],  (const float*)d_in[13] };
    const float* cw[2]   = { (const float*)d_in[5],  (const float*)d_in[14] };
    const float* cb[2]   = { (const float*)d_in[6],  (const float*)d_in[15] };
    const float* xw[2]   = { (const float*)d_in[7],  (const float*)d_in[16] };
    const float* dtw[2]  = { (const float*)d_in[8],  (const float*)d_in[17] };
    const float* dtb[2]  = { (const float*)d_in[9],  (const float*)d_in[18] };
    const float* Alog[2] = { (const float*)d_in[10], (const float*)d_in[19] };
    const float* Dp[2]   = { (const float*)d_in[11], (const float*)d_in[20] };
    const float* outw[2] = { (const float*)d_in[12], (const float*)d_in[21] };
    float* out = (float*)d_out;

    char* ws = (char*)d_ws;
    size_t off = 0;
    auto carve = [&](size_t bytes) -> void* {
        void* p = ws + off;
        off += (bytes + 255) & ~(size_t)255;
        return p;
    };
    float* xn_f32 = (float*)carve((size_t)MTOT * DM * 4);
    f16*   xn_f16 = (f16*)  carve((size_t)MTOT * DM * 2);
    f16*   pin[2], *pout[2];
    float* xz[2], *xc[2], *bc[2];
    f16*   y16[2];
    for (int d = 0; d < 2; ++d) {
        pin[d]  = (f16*)  carve((size_t)DM * NIN * 2);
        pout[d] = (f16*)  carve((size_t)DI * DM * 2);
        xz[d]   = (float*)carve((size_t)MTOT * NIN * 4);
        xc[d]   = (float*)carve((size_t)MTOT * DI * 4);
        bc[d]   = (float*)carve((size_t)MTOT * NPROJ * 4);
        y16[d]  = (f16*)  carve((size_t)MTOT * DI * 2);
    }

    ln_kernel<<<MTOT, 256, 0, stream>>>(x, ln_g, ln_b, xn_f32, xn_f16);

    for (int d = 0; d < 2; ++d) {
        pack_w_kernel<<<2048, 256, 0, stream>>>(inw[d],  pin[d],  DM, NIN);
        pack_w_kernel<<<2048, 256, 0, stream>>>(outw[d], pout[d], DI, DM);
    }

    for (int d = 0; d < 2; ++d) {
        gemm_in_kernel<<<dim3(NIN / 512, MTOT / 16), 256, 0, stream>>>(
            xn_f16, pin[d], xz[d], d);
        size_t nconv = (size_t)MTOT * DI;
        conv_silu_kernel<<<(unsigned)((nconv + 255) / 256), 256, 0, stream>>>(
            xz[d], cw[d], cb[d], xc[d]);
        bcdt_kernel<<<MTOT, 256, 0, stream>>>(xc[d], xw[d], bc[d]);
        scan_kernel<<<dim3(DI / 256, BB), 256, 0, stream>>>(
            bc[d], xc[d], xz[d], dtw[d], dtb[d], Alog[d], Dp[d], y16[d]);
    }

    gemm_out_kernel<<<dim3(2, MTOT / 16), 256, 0, stream>>>(
        y16[0], y16[1], pout[0], pout[1], x, alpha, out);
}
